// MultiHeadSelfAttention_19524921327772
// MI455X (gfx1250) — matrix operand order repaired and checked by reading, not compiled, not run on hardware
//
#include <hip/hip_runtime.h>
#include <stdint.h>
#include <math.h>

// ---------------------------------------------------------------------------
// Multi-head self-attention, CDNA5 (gfx1250) WMMA + TDM implementation.
//   B=2, T=2048, C=1024, H=16, Dk=64.
// f16 WMMA (f32 accum) for all GEMM stages; flash attention keeps the
// per-wave 16x64 score tiles in LDS. Shared tiles (GEMM B-slices, attention
// K/V tiles) are DMA'd into LDS by the Tensor Data Mover, double-buffered
// and pipelined against WMMA via TENSORcnt.
// ---------------------------------------------------------------------------

namespace {
constexpr int kDModel = 1024;
constexpr int kHeads  = 16;
constexpr int kDk     = 64;
constexpr int kSeq    = 2048;
constexpr int kBatch  = 2;
constexpr int kM      = kBatch * kSeq;   // 4096 rows (b*t)
constexpr int kNqkv   = 3 * kDModel;     // 3072
}  // namespace

typedef _Float16 half8 __attribute__((ext_vector_type(8)));
typedef _Float16 v16h  __attribute__((ext_vector_type(16)));
typedef float    v8f   __attribute__((ext_vector_type(8)));
typedef unsigned int u32x4 __attribute__((ext_vector_type(4)));
typedef int i32x4 __attribute__((ext_vector_type(4)));
typedef int i32x8 __attribute__((ext_vector_type(8)));

__device__ __forceinline__ v16h join8(half8 lo, half8 hi) {
  return __builtin_shufflevector(lo, hi, 0, 1, 2, 3, 4, 5, 6, 7,
                                 8, 9, 10, 11, 12, 13, 14, 15);
}
// B fragment (32x16 KxN f16): lane holds column (lane&15); elements 0..7 =
// K[(lane>>4)*8..], elements 8..15 = K[16+(lane>>4)*8..]. p already
// includes the (lane>>4)*8 offset.
__device__ __forceinline__ v16h ldB(const _Float16* p) {
  return join8(*(const half8*)(p), *(const half8*)(p + 16));
}
// A fragment (16x32 MxK f16): lane holds row (lane&15); elements 0..7 =
// K[(lane>>4)*8..], elements 8..15 = K[16+(lane>>4)*8..]. p already
// includes the (lane>>4)*8 offset.
__device__ __forceinline__ v16h ldA(const _Float16* p) {
  return join8(*(const half8*)(p), *(const half8*)(p + 16));
}
__device__ __forceinline__ v8f wmma16(v16h a, v16h b, v8f c) {
  return __builtin_amdgcn_wmma_f32_16x16x32_f16(false, a, false, b,
                                                (short)0, c, false, false);
}

__device__ __forceinline__ uint32_t lds_off(const void* p) {
  // Generic LDS pointers carry the LDS byte offset in the low 32 bits.
  return (uint32_t)(uintptr_t)p;
}

// TDM 2D tile load (f16 elements): tile_x elements per row, tile_y rows,
// row stride stride_x elements, packed contiguously into LDS at lds_addr.
// Descriptor per CDNA5 ISA ch.10.8 (D# groups 0/1; groups 2/3 zero for 2D).
__device__ __forceinline__ void tdm_load_2d_f16(
    uint32_t lds_addr, const _Float16* gptr, uint32_t tile_x, uint32_t tile_y,
    uint32_t tensor_x, uint32_t tensor_y, uint64_t stride_x) {
  uint64_t ga = (uint64_t)(uintptr_t)gptr;
  u32x4 g0;
  g0[0] = 1u;                                   // count=1 (valid), user mode
  g0[1] = lds_addr;                             // lds_addr [63:32]
  g0[2] = (uint32_t)ga;                         // global_addr [95:64]
  g0[3] = (uint32_t)((ga >> 32) & 0x01FFFFFFull) | 0x80000000u;  // type=2
  i32x8 g1;
  g1[0] = (int)(1u << 16);                      // data_size=1 -> 2 bytes
  g1[1] = (int)((tensor_x & 0xFFFFu) << 16);    // tensor_dim0[15:0] @ [63:48]
  g1[2] = (int)(((tensor_x >> 16) & 0xFFFFu) |  // tensor_dim0[31:16]
                ((tensor_y & 0xFFFFu) << 16));  // tensor_dim1[15:0]
  g1[3] = (int)(((tensor_y >> 16) & 0xFFFFu) |  // tensor_dim1[31:16]
                ((tile_x & 0xFFFFu) << 16));    // tile_dim0
  g1[4] = (int)(tile_y & 0xFFFFu);              // tile_dim1; tile_dim2=0
  g1[5] = (int)(uint32_t)stride_x;              // tensor_dim0_stride[31:0]
  g1[6] = (int)(uint32_t)((stride_x >> 32) & 0xFFFFull);
  g1[7] = 0;
  i32x4 z4 = {0, 0, 0, 0};
#if defined(__clang_major__) && (__clang_major__ >= 23)
  i32x8 z8 = {0, 0, 0, 0, 0, 0, 0, 0};
  __builtin_amdgcn_tensor_load_to_lds(g0, g1, z4, z4, z8, 0);
#else
  __builtin_amdgcn_tensor_load_to_lds(g0, g1, z4, z4, 0);
#endif
}

// ---------------------------------------------------------------------------
__global__ void cvt_f32_to_f16(const float* __restrict__ src,
                               _Float16* __restrict__ dst, int n) {
  int i = blockIdx.x * blockDim.x + threadIdx.x;
  int stride = gridDim.x * blockDim.x;
  for (; i < n; i += stride) dst[i] = (_Float16)src[i];
}

// ---------------------------------------------------------------------------
// QKV projection: y[m,n] = sum_k x[m,k]*qkv_w[n,k] + qkv_b[n].
// Workgroup computes 128(M) x 128(N); wave w owns rows m0wg + 16w.
// The shared 128x32 B-slice is TDM-loaded into LDS, double-buffered.
__global__ __launch_bounds__(256) void qkv_gemm_kernel(
    const _Float16* __restrict__ xh, const _Float16* __restrict__ wh,
    const float* __restrict__ bias, _Float16* __restrict__ Qb,
    _Float16* __restrict__ Kb, _Float16* __restrict__ Vt) {
  __shared__ __attribute__((aligned(32))) _Float16 Bsh[2][128 * 32];
  constexpr int tilesN = kNqkv / 128;  // 24
  const int wave = threadIdx.x >> 5;
  const int lane = threadIdx.x & 31;
  const int n0 = (blockIdx.x % tilesN) * 128;
  const int m0 = (blockIdx.x / tilesN) * 128 + wave * 16;
  const int r = lane & 15, hi = lane >> 4;

  if (wave == 0)
    tdm_load_2d_f16(lds_off(&Bsh[0][0]), wh + (size_t)n0 * kDModel, 32, 128,
                    kDModel, kNqkv, kDModel);

  v8f acc[8] = {};
  const _Float16* arow = xh + (size_t)(m0 + r) * kDModel;
  int ibuf = 0;
  for (int k0 = 0; k0 < kDModel; k0 += 32, ibuf ^= 1) {
    if (wave == 0) {
      if (k0 + 32 < kDModel) {
        tdm_load_2d_f16(lds_off(&Bsh[ibuf ^ 1][0]),
                        wh + (size_t)n0 * kDModel + (k0 + 32), 32, 128,
                        kDModel, kNqkv, kDModel);
        __builtin_amdgcn_s_wait_tensorcnt((short)1);
      } else {
        __builtin_amdgcn_s_wait_tensorcnt((short)0);
      }
    }
    __syncthreads();
    v16h a = ldA(arow + k0 + hi * 8);
    v16h b[8];
#pragma unroll
    for (int j = 0; j < 8; ++j)
      b[j] = ldB(&Bsh[ibuf][(16 * j + r) * 32 + hi * 8]);
#pragma unroll
    for (int j = 0; j < 8; ++j) acc[j] = wmma16(a, b[j], acc[j]);
    __syncthreads();
  }

#pragma unroll
  for (int j = 0; j < 8; ++j) {
#pragma unroll
    for (int i = 0; i < 8; ++i) {
      int m = m0 + hi * 8 + i;  // C/D layout: row = vgpr + (lane>>4)*8
      int n = n0 + 16 * j + r;  //             col = lane&15
      float v = acc[j][i] + bias[n];
      int b2 = m >> 11, t = m & (kSeq - 1);
      int sec = n >> 10, w = n & (kDModel - 1);
      int h = w >> 6, d = w & 63;
      size_t bh = (size_t)(b2 * kHeads + h);
      if (sec == 0)
        Qb[(bh * kSeq + t) * kDk + d] = (_Float16)(v * 0.125f);  // 1/sqrt(64)
      else if (sec == 1)
        Kb[(bh * kSeq + t) * kDk + d] = (_Float16)v;
      else
        Vt[(bh * kDk + d) * kSeq + t] = (_Float16)v;  // transposed store
    }
  }
}

// ---------------------------------------------------------------------------
// Flash attention. All 4 waves of a block share one (b,h); the 64x64 K and
// V^T tiles are TDM-staged into double-buffered LDS and shared. Each wave
// owns 16 query rows with online softmax.
__global__ __launch_bounds__(128) void attn_kernel(
    const _Float16* __restrict__ Qb, const _Float16* __restrict__ Kb,
    const _Float16* __restrict__ Vt, _Float16* __restrict__ AO) {
  __shared__ __attribute__((aligned(32))) _Float16 Ksh[2][64 * 64];  // [s][d]
  __shared__ __attribute__((aligned(32))) _Float16 Vsh[2][64 * 64];  // [d][s]
  __shared__ __attribute__((aligned(32))) float    Sbuf[4][16 * 64];
  __shared__ __attribute__((aligned(32))) _Float16 Pbuf[4][16 * 64];
  __shared__ float alphaBuf[4][16];
  __shared__ float linvBuf[4][16];

  const int wave = threadIdx.x >> 5;
  const int lane = threadIdx.x & 31;
  const int bh   = blockIdx.x >> 5;  // 32 blocks per (b,h)
  const int t0   = ((blockIdx.x & 31) * 4 + wave) * 16;
  const int r = lane & 15, hi = lane >> 4;

  const _Float16* Qp = Qb + (size_t)bh * kSeq * kDk;
  const _Float16* Kp = Kb + (size_t)bh * kSeq * kDk;
  const _Float16* Vp = Vt + (size_t)bh * kDk * kSeq;

  if (wave == 0) {
    tdm_load_2d_f16(lds_off(&Ksh[0][0]), Kp, kDk, 64, kDk, kSeq, kDk);
    tdm_load_2d_f16(lds_off(&Vsh[0][0]), Vp, 64, kDk, kSeq, kDk, kSeq);
  }

  const _Float16* qrow = Qp + (size_t)(t0 + r) * kDk;
  v16h qa0 = ldA(qrow + 0 + hi * 8);
  v16h qa1 = ldA(qrow + 32 + hi * 8);

  v8f o[4] = {};
  float m_i = -1e30f, l_i = 0.0f;  // lanes 0..15: stats for row=lane
  int ibuf = 0;

  for (int s0 = 0; s0 < kSeq; s0 += 64, ibuf ^= 1) {
    if (wave == 0) {
      if (s0 + 64 < kSeq) {
        tdm_load_2d_f16(lds_off(&Ksh[ibuf ^ 1][0]),
                        Kp + (size_t)(s0 + 64) * kDk, kDk, 64, kDk, kSeq, kDk);
        tdm_load_2d_f16(lds_off(&Vsh[ibuf ^ 1][0]), Vp + (s0 + 64), 64, kDk,
                        kSeq, kDk, kSeq);
        __builtin_amdgcn_s_wait_tensorcnt((short)2);
      } else {
        __builtin_amdgcn_s_wait_tensorcnt((short)0);
      }
    }
    __syncthreads();  // K/V tile ready

    // S = Q K^T (16 x 64), Q pre-scaled by 1/sqrt(Dk). Batched frag loads.
    v16h kb[8];
#pragma unroll
    for (int j = 0; j < 4; ++j) {
      const _Float16* kr = &Ksh[ibuf][(16 * j + r) * 64 + hi * 8];
      kb[2 * j] = ldB(kr);
      kb[2 * j + 1] = ldB(kr + 32);
    }
#pragma unroll
    for (int j = 0; j < 4; ++j) {
      v8f sc = {};
      sc = wmma16(qa0, kb[2 * j], sc);
      sc = wmma16(qa1, kb[2 * j + 1], sc);
#pragma unroll
      for (int i = 0; i < 8; ++i)
        Sbuf[wave][(hi * 8 + i) * 64 + 16 * j + r] = sc[i];
    }
    __syncthreads();

    // Online softmax: lane t (0..15) owns row t of this wave's tile.
    if (lane < 16) {
      const float* srow = &Sbuf[wave][lane * 64];
      float mt = m_i;
      for (int s = 0; s < 64; ++s) mt = fmaxf(mt, srow[s]);
      float alpha = __expf(m_i - mt);
      float ls = 0.0f;
      _Float16* prow = &Pbuf[wave][lane * 64];
      for (int s = 0; s < 64; ++s) {
        float p = __expf(srow[s] - mt);
        ls += p;
        prow[s] = (_Float16)p;
      }
      l_i = l_i * alpha + ls;
      m_i = mt;
      alphaBuf[wave][lane] = alpha;
    }
    __syncthreads();

    // Rescale accumulators, then O += P V.
    float al[8];
#pragma unroll
    for (int i = 0; i < 8; ++i) al[i] = alphaBuf[wave][hi * 8 + i];
#pragma unroll
    for (int j = 0; j < 4; ++j)
#pragma unroll
      for (int i = 0; i < 8; ++i) o[j][i] *= al[i];

    const _Float16* prow = &Pbuf[wave][r * 64];
    v16h pa0 = ldA(prow + 0 + hi * 8);
    v16h pa1 = ldA(prow + 32 + hi * 8);
    v16h vb[8];
#pragma unroll
    for (int j = 0; j < 4; ++j) {
      const _Float16* vr = &Vsh[ibuf][(16 * j + r) * 64 + hi * 8];
      vb[2 * j] = ldB(vr);
      vb[2 * j + 1] = ldB(vr + 32);
    }
#pragma unroll
    for (int j = 0; j < 4; ++j) {
      o[j] = wmma16(pa0, vb[2 * j], o[j]);
      o[j] = wmma16(pa1, vb[2 * j + 1], o[j]);
    }
    __syncthreads();  // everyone done with this K/V buffer
  }

  if (lane < 16) linvBuf[wave][lane] = 1.0f / l_i;
  __syncthreads();
  float li[8];
#pragma unroll
  for (int i = 0; i < 8; ++i) li[i] = linvBuf[wave][hi * 8 + i];

  const int b = bh >> 4, h = bh & 15;
#pragma unroll
  for (int j = 0; j < 4; ++j)
#pragma unroll
    for (int i = 0; i < 8; ++i) {
      int t = t0 + hi * 8 + i;
      int d = 16 * j + r;
      AO[(size_t)(b * kSeq + t) * kDModel + h * kDk + d] =
          (_Float16)(o[j][i] * li[i]);
    }
}

// ---------------------------------------------------------------------------
// Output projection: out[m,n] = sum_k AO[m,k]*out_w[n,k] + out_b[n] (fp32).
__global__ __launch_bounds__(256) void out_gemm_kernel(
    const _Float16* __restrict__ ah, const _Float16* __restrict__ wh,
    const float* __restrict__ bias, float* __restrict__ out) {
  __shared__ __attribute__((aligned(32))) _Float16 Bsh[2][128 * 32];
  constexpr int tilesN = kDModel / 128;  // 8
  const int wave = threadIdx.x >> 5;
  const int lane = threadIdx.x & 31;
  const int n0 = (blockIdx.x % tilesN) * 128;
  const int m0 = (blockIdx.x / tilesN) * 128 + wave * 16;
  const int r = lane & 15, hi = lane >> 4;

  if (wave == 0)
    tdm_load_2d_f16(lds_off(&Bsh[0][0]), wh + (size_t)n0 * kDModel, 32, 128,
                    kDModel, kDModel, kDModel);

  v8f acc[8] = {};
  const _Float16* arow = ah + (size_t)(m0 + r) * kDModel;
  int ibuf = 0;
  for (int k0 = 0; k0 < kDModel; k0 += 32, ibuf ^= 1) {
    if (wave == 0) {
      if (k0 + 32 < kDModel) {
        tdm_load_2d_f16(lds_off(&Bsh[ibuf ^ 1][0]),
                        wh + (size_t)n0 * kDModel + (k0 + 32), 32, 128,
                        kDModel, kDModel, kDModel);
        __builtin_amdgcn_s_wait_tensorcnt((short)1);
      } else {
        __builtin_amdgcn_s_wait_tensorcnt((short)0);
      }
    }
    __syncthreads();
    v16h a = ldA(arow + k0 + hi * 8);
    v16h b[8];
#pragma unroll
    for (int j = 0; j < 8; ++j)
      b[j] = ldB(&Bsh[ibuf][(16 * j + r) * 32 + hi * 8]);
#pragma unroll
    for (int j = 0; j < 8; ++j) acc[j] = wmma16(a, b[j], acc[j]);
    __syncthreads();
  }

#pragma unroll
  for (int j = 0; j < 8; ++j) {
#pragma unroll
    for (int i = 0; i < 8; ++i) {
      int m = m0 + hi * 8 + i;
      int n = n0 + 16 * j + r;
      out[(size_t)m * kDModel + n] = acc[j][i] + bias[n];
    }
  }
}

// ---------------------------------------------------------------------------
extern "C" void kernel_launch(void* const* d_in, const int* in_sizes, int n_in,
                              void* d_out, int out_size, void* d_ws,
                              size_t ws_size, hipStream_t stream) {
  (void)in_sizes; (void)n_in; (void)out_size; (void)ws_size;
  const float* x     = (const float*)d_in[0];
  const float* qkv_w = (const float*)d_in[1];
  const float* qkv_b = (const float*)d_in[2];
  const float* out_w = (const float*)d_in[3];
  const float* out_b = (const float*)d_in[4];
  float* out = (float*)d_out;

  char* ws = (char*)d_ws;
  const size_t MB = size_t(1) << 20;
  _Float16* xh  = (_Float16*)(ws + 0 * MB);   // 8 MiB  x as f16
  _Float16* wqh = (_Float16*)(ws + 8 * MB);   // 6 MiB  qkv_w f16
  _Float16* owh = (_Float16*)(ws + 14 * MB);  // 2 MiB  out_w f16
  _Float16* Qb  = (_Float16*)(ws + 16 * MB);  // 8 MiB  Q [B,H,T,Dk] (scaled)
  _Float16* Kb  = (_Float16*)(ws + 24 * MB);  // 8 MiB  K [B,H,T,Dk]
  _Float16* Vt  = (_Float16*)(ws + 32 * MB);  // 8 MiB  V^T [B,H,Dk,T]
  _Float16* AO  = (_Float16*)(ws + 40 * MB);  // 8 MiB  attn out [B,T,C]

  cvt_f32_to_f16<<<2048, 256, 0, stream>>>(x, xh, kM * kDModel);
  cvt_f32_to_f16<<<1536, 256, 0, stream>>>(qkv_w, wqh, kNqkv * kDModel);
  cvt_f32_to_f16<<<512, 256, 0, stream>>>(out_w, owh, kDModel * kDModel);

  // (4096/128) * (3072/128) = 32*24 workgroups of 8 waves
  qkv_gemm_kernel<<<768, 256, 0, stream>>>(xh, wqh, qkv_b, Qb, Kb, Vt);
  // 32 (b,h) * 32 query-tile groups, 4 waves each
  attn_kernel<<<1024, 128, 0, stream>>>(Qb, Kb, Vt, AO);
  // (4096/128) * (1024/128) = 32*8 workgroups
  out_gemm_kernel<<<256, 256, 0, stream>>>(AO, owh, out_b, out);
}
